// A2C_DND_14869176778798
// MI455X (gfx1250) — compile-verified, matrix operand order checked by reading, not run on hardware
//
#include <hip/hip_runtime.h>

typedef unsigned short u16;
typedef unsigned int u32;

#define kB 1024
#define kA 4
#define kH 512
#define kDKEY 512
#define kNMEM 65536
#define kOBS 9
#define kE1 64
#define kE2 128
#define kDIN 133
#define kKCAT 672            // 133 + 512, zero-padded to 21*32
#define kGATES 2560
#define kNPART 16
#define kPARTK (kNMEM / kNPART)   // 4096 keys per partition
#define kKT 128                   // keys per iteration
#define kITERS (kPARTK / kKT)     // 32

typedef __attribute__((ext_vector_type(16))) __bf16 bf16x16;
typedef __attribute__((ext_vector_type(8)))  float  f32x8;

__device__ __forceinline__ u16 f2bf(float x) {
  u32 u = __float_as_uint(x);
  u32 r = u + 0x7FFFu + ((u >> 16) & 1u);
  return (u16)(r >> 16);
}

// Load a 16-bit A/B WMMA fragment from a row-major row pointer (row*ld + k0
// already applied). Per ISA layout: lanes 0-15 hold K {0..7, 16..23},
// lanes 16-31 hold K {8..15, 24..31}  -> two contiguous 16B chunks per lane.
__device__ __forceinline__ bf16x16 load_frag(const u16* p, int half) {
  union { bf16x16 v; uint4 q[2]; } f;
  f.q[0] = *reinterpret_cast<const uint4*>(p + half * 8);
  f.q[1] = *reinterpret_cast<const uint4*>(p + 16 + half * 8);
  return f.v;
}

__device__ __forceinline__ f32x8 wmma_bf16(bf16x16 a, bf16x16 b, f32x8 c) {
  return __builtin_amdgcn_wmma_f32_16x16x32_bf16(false, a, false, b, (short)0, c,
                                                 false, false);
}

__device__ __forceinline__ float sigmf(float x) { return 1.0f / (1.0f + __expf(-x)); }

// ---------------------------------------------------------------------------
// Prep: fp32 row -> bf16 row + squared L2 norm (used for cue and dnd_keys)
// ---------------------------------------------------------------------------
__global__ __launch_bounds__(128) void prep_rows_bf16(const float* __restrict__ src,
                                                      u16* __restrict__ dst,
                                                      float* __restrict__ sqn) {
  const int row = blockIdx.x;
  const int tid = threadIdx.x;
  const float4 v = reinterpret_cast<const float4*>(src + (size_t)row * 512)[tid];
  uint2 pk;
  pk.x = (u32)f2bf(v.x) | ((u32)f2bf(v.y) << 16);
  pk.y = (u32)f2bf(v.z) | ((u32)f2bf(v.w) << 16);
  reinterpret_cast<uint2*>(dst + (size_t)row * 512)[tid] = pk;
  float acc = v.x * v.x + v.y * v.y + v.z * v.z + v.w * v.w;
#pragma unroll
  for (int m = 1; m < 32; m <<= 1) acc += __shfl_xor(acc, m, 32);
  __shared__ float red[4];
  if ((tid & 31) == 0) red[tid >> 5] = acc;
  __syncthreads();
  if (tid == 0) sqn[row] = red[0] + red[1] + red[2] + red[3];
}

// ---------------------------------------------------------------------------
// Prep: dnd_vals [NMEM,H] fp32 -> valsT [H,NMEM] bf16 (32x32 LDS tile transpose)
// ---------------------------------------------------------------------------
__global__ __launch_bounds__(256) void prep_vals_transpose(const float* __restrict__ vals,
                                                           u16* __restrict__ valsT) {
  __shared__ u16 t[32][33];
  const int r0 = blockIdx.x * 32;  // NMEM dim
  const int c0 = blockIdx.y * 32;  // H dim
  const int tx = threadIdx.x, ty = threadIdx.y;
#pragma unroll
  for (int i = 0; i < 4; ++i)
    t[ty + 8 * i][tx] = f2bf(vals[(size_t)(r0 + ty + 8 * i) * kH + c0 + tx]);
  __syncthreads();
#pragma unroll
  for (int i = 0; i < 4; ++i)
    valsT[(size_t)(c0 + ty + 8 * i) * kNMEM + r0 + tx] = t[tx][ty + 8 * i];
}

// ---------------------------------------------------------------------------
// Prep: wcat[g] = [w_ih[g] | w_hh[g] | zeros]  (bf16, K = 672)
// ---------------------------------------------------------------------------
__global__ __launch_bounds__(128) void prep_wcat(const float* __restrict__ w_ih,
                                                 const float* __restrict__ w_hh,
                                                 u16* __restrict__ wcat) {
  const int g = blockIdx.x;
  const int tid = threadIdx.x;
  u16* row = wcat + (size_t)g * kKCAT;
  for (int j = tid; j < kDIN; j += 128) row[j] = f2bf(w_ih[(size_t)g * kDIN + j]);
  for (int j = tid; j < kH; j += 128) row[kDIN + j] = f2bf(w_hh[(size_t)g * kH + j]);
  if (tid < kKCAT - (kDIN + kH)) row[kDIN + kH + tid] = (u16)0;
}

// ---------------------------------------------------------------------------
// Encoder MLP + build xh = [feats | p_action | p_reward | h0 | pad] bf16
// ---------------------------------------------------------------------------
__global__ __launch_bounds__(128) void encoder_xh(
    const float* __restrict__ obs, const float* __restrict__ p_action,
    const float* __restrict__ p_reward, const float* __restrict__ h0,
    const float* __restrict__ w1, const float* __restrict__ b1,
    const float* __restrict__ w2, const float* __restrict__ b2,
    float* __restrict__ out_feats, u16* __restrict__ xh) {
  __shared__ float s_obs[kOBS];
  __shared__ float s_h1[kE1];
  const int b = blockIdx.x;
  const int tid = threadIdx.x;
  if (tid < kOBS) s_obs[tid] = obs[b * kOBS + tid];
  __syncthreads();
  if (tid < kE1) {
    float a = b1[tid];
#pragma unroll
    for (int k = 0; k < kOBS; ++k) a += s_obs[k] * w1[tid * kOBS + k];
    s_h1[tid] = fmaxf(a, 0.0f);
  }
  __syncthreads();
  u16* row = xh + (size_t)b * kKCAT;
  {
    float a = b2[tid];
    for (int k = 0; k < kE1; ++k) a += s_h1[k] * w2[tid * kE1 + k];
    const float f = fmaxf(a, 0.0f);
    out_feats[b * kE2 + tid] = f;
    row[tid] = f2bf(f);
  }
  if (tid < kA) row[kE2 + tid] = f2bf(p_action[b * kA + tid]);
  if (tid == 4) row[kE2 + kA] = f2bf(p_reward[b]);
  for (int j = tid; j < kH; j += 128) row[kDIN + j] = f2bf(h0[(size_t)b * kH + j]);
  if (tid < kKCAT - (kDIN + kH)) row[kDIN + kH + tid] = (u16)0;
}

// ---------------------------------------------------------------------------
// Flash DND attention over one NMEM partition.
//   grid = (B/16, NPART), block = 256 (8 waves).
//   Per iteration (128 keys): each wave computes one 16x16 S tile over K=512
//   (16 bf16 WMMAs), online softmax shared via LDS, then each wave runs
//   P[16,128] @ V[128, its 64 H-cols] (16 bf16 WMMAs) into persistent C frags.
// ---------------------------------------------------------------------------
#define LQS 520   // cue tile LDS row stride (u16), 1040B = 65*16 (bank spread)
#define LPS 136   // P tile LDS row stride (u16), 272B = 17*16

__global__ __launch_bounds__(256) void dnd_attention(
    const u16* __restrict__ cueb, const float* __restrict__ qsqn,
    const u16* __restrict__ keysb, const float* __restrict__ ksqn,
    const u16* __restrict__ valsT,
    float* __restrict__ part_out, float* __restrict__ stat_m,
    float* __restrict__ stat_l) {
  __shared__ __align__(16) u16 lq[16 * LQS];
  __shared__ __align__(16) u16 lp[16 * LPS];
  __shared__ float wmax[8][16];
  __shared__ float wsum[8][16];
  __shared__ float rm[16], rl[16], rscale[16], rqn[16];

  const int tid = threadIdx.x;
  const int wave = tid >> 5;
  const int lane = tid & 31;
  const int half = lane >> 4;
  const int ln = lane & 15;
  const int qbase = blockIdx.x * 16;
  const int part = blockIdx.y;
  const int kbase = part * kPARTK;

  // stage cue tile (16 x 512 bf16) into LDS, 16B per thread per step
  for (int i = tid; i < 1024; i += 256) {
    const int r = i >> 6, c = i & 63;
    *reinterpret_cast<uint4*>(lq + r * LQS + c * 8) =
        *reinterpret_cast<const uint4*>(cueb + (size_t)(qbase + r) * kDKEY + c * 8);
  }
  if (tid < 16) {
    rm[tid] = -3.0e38f;
    rl[tid] = 0.0f;
    rqn[tid] = qsqn[qbase + tid];
  }
  __syncthreads();

  float qn8[8];
#pragma unroll
  for (int r = 0; r < 8; ++r) qn8[r] = rqn[r + 8 * half];

  f32x8 accV[4];
#pragma unroll
  for (int t = 0; t < 4; ++t)
#pragma unroll
    for (int r = 0; r < 8; ++r) accV[t][r] = 0.0f;

  const u16* qrow = lq + ln * LQS;

  for (int it = 0; it < kITERS; ++it) {
    const int kt = kbase + it * kKT;
    const u16* krow = keysb + (size_t)(kt + wave * 16 + ln) * kDKEY;
    if (it + 1 < kITERS)
      __builtin_prefetch(keysb + (size_t)(kt + kKT + wave * 16 + ln) * kDKEY, 0, 1);

    // ---- S tile: this wave owns keys [kt + wave*16, +16), dot over K=512
    f32x8 accS;
#pragma unroll
    for (int r = 0; r < 8; ++r) accS[r] = 0.0f;
#pragma unroll 4
    for (int ks = 0; ks < 16; ++ks) {
      bf16x16 a = load_frag(qrow + ks * 32, half);
      bf16x16 b = load_frag(krow + ks * 32, half);
      accS = wmma_bf16(a, b, accS);
    }

    const float kn = ksqn[kt + wave * 16 + ln];
    float sim[8];
#pragma unroll
    for (int r = 0; r < 8; ++r)
      sim[r] = -sqrtf(fmaxf(qn8[r] + kn - 2.0f * accS[r], 1e-12f));

    // per-row max within the 16-lane half (C-frag row r + 8*half)
#pragma unroll
    for (int r = 0; r < 8; ++r) {
      float v = sim[r];
      v = fmaxf(v, __shfl_xor(v, 1, 32));
      v = fmaxf(v, __shfl_xor(v, 2, 32));
      v = fmaxf(v, __shfl_xor(v, 4, 32));
      v = fmaxf(v, __shfl_xor(v, 8, 32));
      if (ln == 0) wmax[wave][r + 8 * half] = v;
    }
    __syncthreads();  // A: wmax visible
    if (tid < 16) {
      const float mo = rm[tid];
      float mx = mo;
#pragma unroll
      for (int w = 0; w < 8; ++w) mx = fmaxf(mx, wmax[w][tid]);
      rscale[tid] = __expf(mo - mx);
      rm[tid] = mx;
    }
    __syncthreads();  // B: new rm/rscale visible

    // P = exp(sim - m) -> bf16 LDS tile; per-row partial sums
#pragma unroll
    for (int r = 0; r < 8; ++r) {
      const float p = __expf(sim[r] - rm[r + 8 * half]);
      lp[(r + 8 * half) * LPS + wave * 16 + ln] = f2bf(p);
      float s = p;
      s += __shfl_xor(s, 1, 32);
      s += __shfl_xor(s, 2, 32);
      s += __shfl_xor(s, 4, 32);
      s += __shfl_xor(s, 8, 32);
      if (ln == 0) wsum[wave][r + 8 * half] = s;
    }
    float sc8[8];
#pragma unroll
    for (int r = 0; r < 8; ++r) sc8[r] = rscale[r + 8 * half];
    __syncthreads();  // C: lp + wsum visible
    if (tid < 16) {
      float s = 0.0f;
#pragma unroll
      for (int w = 0; w < 8; ++w) s += wsum[w][tid];
      rl[tid] = rl[tid] * rscale[tid] + s;
    }

    // rescale persistent accumulators, then P @ V for this wave's 64 H cols
#pragma unroll
    for (int t = 0; t < 4; ++t)
#pragma unroll
      for (int r = 0; r < 8; ++r) accV[t][r] *= sc8[r];

    const u16* prow = lp + ln * LPS;
#pragma unroll
    for (int ksp = 0; ksp < 4; ++ksp) {
      bf16x16 ap = load_frag(prow + ksp * 32, half);
#pragma unroll
      for (int t = 0; t < 4; ++t) {
        const u16* vb =
            valsT + (size_t)(wave * 64 + t * 16 + ln) * kNMEM + kt + ksp * 32;
        bf16x16 bv = load_frag(vb, half);
        accV[t] = wmma_bf16(ap, bv, accV[t]);
      }
    }
    __syncthreads();  // D: done reading lp before next iteration rewrites it
  }

  // write partial numerators + stats for cross-partition combine
#pragma unroll
  for (int t = 0; t < 4; ++t)
#pragma unroll
    for (int r = 0; r < 8; ++r)
      part_out[((size_t)part * kB + qbase + r + 8 * half) * kH + wave * 64 +
               t * 16 + ln] = accV[t][r];
  if (tid < 16) {
    stat_m[part * kB + qbase + tid] = rm[tid];
    stat_l[part * kB + qbase + tid] = rl[tid];
  }
}

// ---------------------------------------------------------------------------
// Combine partitions with log-sum-exp rescale: m_t[b,h]
// ---------------------------------------------------------------------------
__global__ __launch_bounds__(256) void dnd_combine(const float* __restrict__ part,
                                                   const float* __restrict__ sm,
                                                   const float* __restrict__ sl,
                                                   float* __restrict__ mt) {
  const int b = blockIdx.x;
  const int tid = threadIdx.x;
  __shared__ float wgt[kNPART];
  __shared__ float sden;
  if (tid == 0) {
    float M = -3.0e38f;
    for (int p = 0; p < kNPART; ++p) M = fmaxf(M, sm[p * kB + b]);
    float den = 0.0f;
    for (int p = 0; p < kNPART; ++p) {
      const float w = __expf(sm[p * kB + b] - M);
      wgt[p] = w;
      den += sl[p * kB + b] * w;
    }
    sden = den;
  }
  __syncthreads();
  const float inv = 1.0f / sden;
  for (int h = tid; h < kH; h += 256) {
    float num = 0.0f;
    for (int p = 0; p < kNPART; ++p)
      num += part[((size_t)p * kB + b) * kH + h] * wgt[p];
    mt[(size_t)b * kH + h] = num * inv;
  }
}

// ---------------------------------------------------------------------------
// LSTM pre-activation GEMM: z[1024,2560] = xh[1024,672] @ wcat[2560,672]^T
//   grid (64, 40), block 128 (4 waves), one 16x16 tile per wave, K = 21*32.
// ---------------------------------------------------------------------------
__global__ __launch_bounds__(128) void lstm_gemm(const u16* __restrict__ xh,
                                                 const u16* __restrict__ wcat,
                                                 float* __restrict__ z) {
  const int wave = threadIdx.x >> 5;
  const int lane = threadIdx.x & 31;
  const int half = lane >> 4;
  const int ln = lane & 15;
  const int mt = blockIdx.x;
  const int nt = blockIdx.y * 4 + wave;
  const u16* arow = xh + (size_t)(mt * 16 + ln) * kKCAT;
  const u16* brow = wcat + (size_t)(nt * 16 + ln) * kKCAT;
  f32x8 acc;
#pragma unroll
  for (int r = 0; r < 8; ++r) acc[r] = 0.0f;
#pragma unroll 3
  for (int ks = 0; ks < 21; ++ks) {
    bf16x16 a = load_frag(arow + ks * 32, half);
    bf16x16 b = load_frag(brow + ks * 32, half);
    acc = wmma_bf16(a, b, acc);
  }
#pragma unroll
  for (int r = 0; r < 8; ++r)
    z[(size_t)(mt * 16 + r + 8 * half) * kGATES + nt * 16 + ln] = acc[r];
}

// ---------------------------------------------------------------------------
// LSTM elementwise: gates + memory reinstatement -> h_new, c_new
// ---------------------------------------------------------------------------
__global__ __launch_bounds__(256) void lstm_elem(
    const float* __restrict__ z, const float* __restrict__ b_ih,
    const float* __restrict__ b_hh, const float* __restrict__ c0,
    const float* __restrict__ mt, float* __restrict__ out_h,
    float* __restrict__ out_c) {
  const int idx = blockIdx.x * 256 + threadIdx.x;  // = b*512 + u
  const int b = idx >> 9;
  const int u = idx & 511;
  const float* zb = z + (size_t)b * kGATES;
  float g[5];
#pragma unroll
  for (int k = 0; k < 5; ++k)
    g[k] = zb[k * kH + u] + b_ih[k * kH + u] + b_hh[k * kH + u];
  const float ig = sigmf(g[0]);
  const float fg = sigmf(g[1]);
  const float gg = tanhf(g[2]);
  const float og = sigmf(g[3]);
  const float rg = sigmf(g[4]);
  const float cn = fg * c0[idx] + ig * gg + rg * mt[idx];
  out_c[idx] = cn;
  out_h[idx] = og * tanhf(cn);
}

// ---------------------------------------------------------------------------
// Actor/critic heads: one wave per batch row
// ---------------------------------------------------------------------------
__global__ __launch_bounds__(128) void heads_kernel(
    const float* __restrict__ h, const float* __restrict__ aw,
    const float* __restrict__ ab, const float* __restrict__ cw,
    const float* __restrict__ cb, float* __restrict__ logits,
    float* __restrict__ value) {
  const int wave = threadIdx.x >> 5;
  const int lane = threadIdx.x & 31;
  const int b = blockIdx.x * 4 + wave;
  const float* hr = h + (size_t)b * kH;
  float a0 = 0, a1 = 0, a2 = 0, a3 = 0, v = 0;
  for (int j = lane; j < kH; j += 32) {
    const float x = hr[j];
    a0 += x * aw[j];
    a1 += x * aw[kH + j];
    a2 += x * aw[2 * kH + j];
    a3 += x * aw[3 * kH + j];
    v += x * cw[j];
  }
#pragma unroll
  for (int m = 1; m < 32; m <<= 1) {
    a0 += __shfl_xor(a0, m, 32);
    a1 += __shfl_xor(a1, m, 32);
    a2 += __shfl_xor(a2, m, 32);
    a3 += __shfl_xor(a3, m, 32);
    v += __shfl_xor(v, m, 32);
  }
  if (lane == 0) {
    logits[b * 4 + 0] = a0 + ab[0];
    logits[b * 4 + 1] = a1 + ab[1];
    logits[b * 4 + 2] = a2 + ab[2];
    logits[b * 4 + 3] = a3 + ab[3];
    value[b] = v + cb[0];
  }
}

// ---------------------------------------------------------------------------
extern "C" void kernel_launch(void* const* d_in, const int* in_sizes, int n_in,
                              void* d_out, int out_size, void* d_ws, size_t ws_size,
                              hipStream_t stream) {
  (void)in_sizes; (void)n_in; (void)out_size; (void)ws_size;

  const float* obs      = (const float*)d_in[0];
  const float* p_action = (const float*)d_in[1];
  const float* p_reward = (const float*)d_in[2];
  const float* h0       = (const float*)d_in[3];
  const float* c0       = (const float*)d_in[4];
  const float* cue      = (const float*)d_in[5];
  const float* enc_w1   = (const float*)d_in[6];
  const float* enc_b1   = (const float*)d_in[7];
  const float* enc_w2   = (const float*)d_in[8];
  const float* enc_b2   = (const float*)d_in[9];
  const float* w_ih     = (const float*)d_in[10];
  const float* w_hh     = (const float*)d_in[11];
  const float* b_ih     = (const float*)d_in[12];
  const float* b_hh     = (const float*)d_in[13];
  const float* actor_w  = (const float*)d_in[14];
  const float* actor_b  = (const float*)d_in[15];
  const float* critic_w = (const float*)d_in[16];
  const float* critic_b = (const float*)d_in[17];
  const float* dnd_keys = (const float*)d_in[18];
  const float* dnd_vals = (const float*)d_in[19];

  float* out        = (float*)d_out;
  float* out_logits = out;                       // [1024,4]
  float* out_value  = out + 4096;                // [1024,1]
  float* out_h      = out + 5120;                // [1,1024,512]
  float* out_c      = out + 529408;              // [1,1024,512]
  float* out_feats  = out + 1053696;             // [1024,128]

  char* base = (char*)d_ws;
  size_t off = 0;
  auto take = [&](size_t bytes) -> void* {
    void* p = base + off;
    off = (off + bytes + 255) & ~(size_t)255;
    return p;
  };
  u16*   ws_keysb = (u16*)take((size_t)kNMEM * kDKEY * 2);   // 64 MB
  u16*   ws_valsT = (u16*)take((size_t)kNMEM * kH * 2);      // 64 MB
  u16*   ws_cueb  = (u16*)take((size_t)kB * kDKEY * 2);      // 1 MB
  float* ws_ksqn  = (float*)take((size_t)kNMEM * 4);
  float* ws_qsqn  = (float*)take((size_t)kB * 4);
  float* ws_part  = (float*)take((size_t)kNPART * kB * kH * 4);  // 32 MB
  float* ws_sm    = (float*)take((size_t)kNPART * kB * 4);
  float* ws_sl    = (float*)take((size_t)kNPART * kB * 4);
  float* ws_mt    = (float*)take((size_t)kB * kH * 4);
  u16*   ws_xh    = (u16*)take((size_t)kB * kKCAT * 2);
  u16*   ws_wcat  = (u16*)take((size_t)kGATES * kKCAT * 2);
  float* ws_z     = (float*)take((size_t)kB * kGATES * 4);   // 10 MB

  // prep passes (memory-bound, ~384MB total traffic)
  prep_rows_bf16<<<kNMEM, 128, 0, stream>>>(dnd_keys, ws_keysb, ws_ksqn);
  prep_rows_bf16<<<kB, 128, 0, stream>>>(cue, ws_cueb, ws_qsqn);
  prep_vals_transpose<<<dim3(kNMEM / 32, kH / 32), dim3(32, 8), 0, stream>>>(
      dnd_vals, ws_valsT);
  prep_wcat<<<kGATES, 128, 0, stream>>>(w_ih, w_hh, ws_wcat);
  encoder_xh<<<kB, 128, 0, stream>>>(obs, p_action, p_reward, h0, enc_w1, enc_b1,
                                     enc_w2, enc_b2, out_feats, ws_xh);

  // flash DND attention (dominant compute: ~137 GFLOP of bf16 WMMA)
  dnd_attention<<<dim3(kB / 16, kNPART), 256, 0, stream>>>(
      ws_cueb, ws_qsqn, ws_keysb, ws_ksqn, ws_valsT, ws_part, ws_sm, ws_sl);
  dnd_combine<<<kB, 256, 0, stream>>>(ws_part, ws_sm, ws_sl, ws_mt);

  // LSTM
  lstm_gemm<<<dim3(kB / 16, kGATES / 64), 128, 0, stream>>>(ws_xh, ws_wcat, ws_z);
  lstm_elem<<<(kB * kH) / 256, 256, 0, stream>>>(ws_z, b_ih, b_hh, c0, ws_mt,
                                                 out_h, out_c);

  // heads
  heads_kernel<<<kB / 4, 128, 0, stream>>>(out_h, actor_w, actor_b, critic_w,
                                           critic_b, out_logits, out_value);
}
